// MixTemporalGNN_30846455120314
// MI455X (gfx1250) — compile-verified, matrix operand order checked by reading, not run on hardware
//
#include <hip/hip_runtime.h>
#include <hip/hip_bf16.h>

// MixTemporalGNN for MI455X (gfx1250, wave32).
// - bf16 WMMA (v_wmma_f32_16x16x32_bf16) for fused rst = [X|mean] @ [Ws;Wn] + b
// - async tensor path: GLOBAL_LOAD_ASYNC_TO_LDS_B128 stages bf16 tiles
// - edge aggregation: native global_atomic_add_f32 (bandwidth-dominant stage)

#define NN    100000
#define EE    800000
#define GG    64
#define EMBD  64
#define HF    128
#define EPSV  1e-5f

typedef __attribute__((ext_vector_type(16))) __bf16 v16bf;
typedef __attribute__((ext_vector_type(8)))  float  v8f;

__device__ __forceinline__ void atomAdd(float* p, float v) {
  __hip_atomic_fetch_add(p, v, __ATOMIC_RELAXED, __HIP_MEMORY_SCOPE_AGENT);
}

// f32 -> bf16, round-to-nearest-even
__device__ __forceinline__ unsigned short f2bf(float f) {
  unsigned int u = __builtin_bit_cast(unsigned int, f);
  u += 0x7FFFu + ((u >> 16) & 1u);
  return (unsigned short)(u >> 16);
}

// 16-byte async copy global -> LDS (ASYNCcnt-tracked, CDNA5 ISA)
__device__ __forceinline__ void async_cp16(const unsigned short* g, unsigned short* l) {
  asm volatile("global_load_async_to_lds_b128 %0, %1, off"
               :: "v"((unsigned)(unsigned long long)l),
                  "v"((unsigned long long)g)
               : "memory");
}
__device__ __forceinline__ void async_wait0() {
  asm volatile("s_wait_asynccnt 0x0" ::: "memory");
}

__global__ void k_zero(float* p, long n) {
  long i = (long)blockIdx.x * blockDim.x + threadIdx.x;
  long s = (long)gridDim.x * blockDim.x;
  for (; i < n; i += s) p[i] = 0.f;
}

// h[node,0:64] = emb[feat[node],:]  (f32 for aggregation + bf16 for WMMA)
__global__ void k_embed(const int* __restrict__ feat, const float* __restrict__ emb,
                        float* __restrict__ h, unsigned short* __restrict__ hbf, int n) {
  int i = blockIdx.x * blockDim.x + threadIdx.x;
  if (i >= n * EMBD) return;
  int node = i >> 6, c = i & 63;
  float v = emb[(size_t)feat[node] * EMBD + c];
  h[(size_t)node * HF + c]   = v;
  hbf[(size_t)node * HF + c] = f2bf(v);
}

__global__ void k_count(const int* __restrict__ gid, float* __restrict__ cnt, int n) {
  int i = blockIdx.x * blockDim.x + threadIdx.x;
  if (i < n) atomAdd(cnt + gid[i], 1.f);
}

// Edge scatter-add: msum[dst] += h[src]; deg[dst] += 1
__global__ void k_agg(const float* __restrict__ h, const int* __restrict__ src,
                      const int* __restrict__ dst, float* __restrict__ msum,
                      float* __restrict__ deg, int ne, int din) {
  int e = blockIdx.x * blockDim.x + threadIdx.x;
  if (e >= ne) return;
  int s = src[e], d = dst[e];
  const float* hs = h + (size_t)s * HF;
  float* md = msum + (size_t)d * HF;
  __builtin_prefetch(hs, 0, 1);   // global_prefetch_b8
  for (int c = 0; c < din; c += 4) {
    float4 v = *(const float4*)(hs + c);
    atomAdd(md + c + 0, v.x);
    atomAdd(md + c + 1, v.y);
    atomAdd(md + c + 2, v.z);
    atomAdd(md + c + 3, v.w);
  }
  atomAdd(deg + d, 1.f);
}

// Mbf = bf16(msum / max(deg,1))
__global__ void k_mean_bf(const float* __restrict__ msum, const float* __restrict__ deg,
                          unsigned short* __restrict__ Mbf, int n, int din) {
  long i = (long)blockIdx.x * blockDim.x + threadIdx.x;
  if (i >= (long)n * din) return;
  int node = (int)(i / din), c = (int)(i - (long)node * din);
  float dg = deg[node]; dg = dg > 1.f ? dg : 1.f;
  Mbf[(size_t)node * HF + c] = f2bf(msum[(size_t)node * HF + c] / dg);
}

// Pack [Ws ; Wn] (2*din x 128, f32) into column-major bf16 Wc[col*256 + k]
__global__ void k_packw(const float* __restrict__ Ws, const float* __restrict__ Wn,
                        unsigned short* __restrict__ Wc, int din) {
  int i = blockIdx.x * blockDim.x + threadIdx.x;
  int kt = 2 * din;
  if (i >= kt * HF) return;
  int k = i % kt, c = i / kt;
  float v = (k < din) ? Ws[(size_t)k * HF + c] : Wn[(size_t)(k - din) * HF + c];
  Wc[(size_t)c * 256 + k] = f2bf(v);
}

// Fused SAGE GEMM: out = PReLU([X|M] @ Wc + b); bf16 WMMA, fp32 accumulate.
// Tiles staged via async global->LDS byte copy; BN stats reduced in epilogue.
template<int DIN>
__global__ __launch_bounds__(256) void k_sage_gemm(
    const unsigned short* __restrict__ Xbf, const unsigned short* __restrict__ Mbf,
    const unsigned short* __restrict__ Wc,
    const float* __restrict__ bias, const float* __restrict__ alpha,
    float* __restrict__ colsum, float* __restrict__ colsq,
    float* __restrict__ out, int n) {
  __shared__ unsigned short ldsX[64 * 136];   // padded stride (bank spread)
  __shared__ unsigned short ldsM[64 * 136];
  const int tid  = threadIdx.x;
  const int row0 = blockIdx.x * 64;

  // Stage 64 x DIN bf16 tiles of X and M with async LDS copies (16B chunks).
  constexpr int CPR = DIN / 8;        // 16B chunks per row
  constexpr int CH  = 64 * CPR;       // chunks per tile
  #pragma unroll
  for (int ch = 0; ch < CH / 256; ++ch) {
    int idx = ch * 256 + tid;
    int r = idx / CPR;
    int k = (idx - r * CPR) * 8;
    size_t goff = (size_t)(row0 + r) * HF + k;   // tail over-read: pad rows only
    async_cp16(Xbf + goff, &ldsX[r * 136 + k]);
    async_cp16(Mbf + goff, &ldsM[r * 136 + k]);
  }
  async_wait0();
  __syncthreads();

  const int lane = tid & 31;
  const int wave = tid >> 5;
  const int c0   = wave * 16;           // 8 waves x 16 cols = 128 cols
  const int nIdx = lane & 15;
  const int half = lane >> 4;
  const int col  = c0 + nIdx;

  union AB { v16bf v; unsigned int u[8]; };
  union C8 { v8f v; float f[8]; };
  C8 acc[4];
  #pragma unroll
  for (int rt = 0; rt < 4; ++rt)
    #pragma unroll
    for (int j = 0; j < 8; ++j) acc[rt].f[j] = 0.f;

  #pragma unroll
  for (int k0 = 0; k0 < 2 * DIN; k0 += 32) {
    // B fragment (32x16 bf16): col = lane&15, K split by lane half (ISA 7.12.2)
    AB B;
    const unsigned short* wb = Wc + (size_t)col * 256 + k0 + half * 8;
    #pragma unroll
    for (int j = 0; j < 4; ++j) {
      B.u[j]     = *(const unsigned int*)(wb + 2 * j);
      B.u[4 + j] = *(const unsigned int*)(wb + 16 + 2 * j);
    }
    // A window lies fully in X or M half (DIN % 32 == 0)
    const unsigned short* abase = (k0 < DIN) ? ldsX : ldsM;
    const int kl = (k0 < DIN) ? k0 : (k0 - DIN);
    #pragma unroll
    for (int rt = 0; rt < 4; ++rt) {
      AB A;   // 16x32 bf16: row = lane&15; v0..3 K=half*8+2j, v4..7 K=+16
      const unsigned short* ar = abase + (rt * 16 + nIdx) * 136 + kl + half * 8;
      #pragma unroll
      for (int j = 0; j < 4; ++j) {
        A.u[j]     = *(const unsigned int*)(ar + 2 * j);
        A.u[4 + j] = *(const unsigned int*)(ar + 16 + 2 * j);
      }
      acc[rt].v = __builtin_amdgcn_wmma_f32_16x16x32_bf16(
          false, A.v, false, B.v, (short)0, acc[rt].v, false, false);
    }
  }

  // Epilogue: bias + PReLU, store rst, reduce BN stats in-register, 2 atomics
  float bcol = bias[col];
  float acol = alpha[col];
  float s = 0.f, s2 = 0.f;
  #pragma unroll
  for (int rt = 0; rt < 4; ++rt) {
    #pragma unroll
    for (int v = 0; v < 8; ++v) {       // C/D layout: M = v + 8*half, N = lane&15
      int row = row0 + rt * 16 + v + 8 * half;
      if (row < n) {
        float x = acc[rt].f[v] + bcol;
        x = x > 0.f ? x : acol * x;
        out[(size_t)row * HF + col] = x;
        s += x; s2 += x * x;
      }
    }
  }
  s  += __shfl_xor(s, 16, 32);          // combine lane halves (same column)
  s2 += __shfl_xor(s2, 16, 32);
  if (half == 0) {
    atomAdd(&colsum[col], s);
    atomAdd(&colsq[col],  s2);
  }
}

__global__ void k_bnfin(const float* __restrict__ colsum, const float* __restrict__ colsq,
                        const float* __restrict__ gamma, const float* __restrict__ beta,
                        float* __restrict__ scale, float* __restrict__ shift, int n) {
  int c = threadIdx.x;
  if (c >= HF) return;
  float inv = 1.f / (float)n;
  float mu  = colsum[c] * inv;
  float var = colsq[c] * inv - mu * mu;
  float sc  = gamma[c] * rsqrtf(var + EPSV);
  scale[c] = sc;
  shift[c] = beta[c] - mu * sc;
}

// BN apply in place (f32 + bf16 mirror) + per-graph segment accumulation
__global__ void k_bn_apply(float* __restrict__ h, unsigned short* __restrict__ hbf,
                           const int* __restrict__ gid, float* __restrict__ gsum,
                           const float* __restrict__ scale, const float* __restrict__ shift,
                           int n, int outbase) {
  long i = (long)blockIdx.x * blockDim.x + threadIdx.x;
  if (i >= (long)n * HF) return;
  int node = (int)(i >> 7), c = (int)(i & 127);
  float v = h[i] * scale[c] + shift[c];
  h[i]   = v;
  hbf[i] = f2bf(v);
  atomAdd(&gsum[(size_t)gid[node] * 1536 + outbase + c], v);
}

__global__ void k_final(const float* __restrict__ gsum, const float* __restrict__ gcnt,
                        float* __restrict__ out) {
  int i = blockIdx.x * blockDim.x + threadIdx.x;
  if (i >= GG * 1536) return;
  int g = i / 1536, col = i - g * 1536;
  int t = col / 512;                       // 0=h, 1=p, 2=hp
  float c = gcnt[t * GG + g]; c = c > 1.f ? c : 1.f;
  out[i] = gsum[i] / c;
}

extern "C" void kernel_launch(void* const* d_in, const int* in_sizes, int n_in,
                              void* d_out, int out_size, void* d_ws, size_t ws_size,
                              hipStream_t stream) {
  (void)in_sizes; (void)n_in; (void)out_size; (void)ws_size;
  const int* feat[3] = {(const int*)d_in[0], (const int*)d_in[1], (const int*)d_in[2]};
  const int* src [3] = {(const int*)d_in[3], (const int*)d_in[5], (const int*)d_in[7]};
  const int* dst [3] = {(const int*)d_in[4], (const int*)d_in[6], (const int*)d_in[8]};
  const int* gid [3] = {(const int*)d_in[9], (const int*)d_in[10], (const int*)d_in[11]};
  // params pytree, sorted keys: emb_h, emb_hp, emb_p, layers[l]{h,hp,p}{Wn,Ws,alpha,b,beta,gamma}
  const float* emb[3] = {(const float*)d_in[12], (const float*)d_in[14], (const float*)d_in[13]};
  const int slot[3] = {0, 2, 1};   // type order (h,p,hp) -> sorted (h,hp,p)

  // Workspace layout (float units)
  float* W = (float*)d_ws;
  size_t off = 0;
  float* hf[3];
  for (int t = 0; t < 3; ++t) { hf[t] = W + off; off += (size_t)NN * HF; }
  unsigned short* hbf[3];
  for (int t = 0; t < 3; ++t) { hbf[t] = (unsigned short*)(W + off); off += (size_t)NN * HF / 2; }
  unsigned short* Mbf = (unsigned short*)(W + off); off += (size_t)NN * HF / 2;
  float* msum = W + off; off += (size_t)NN * HF;   // msum + deg contiguous
  float* deg  = W + off; off += NN;
  unsigned short* Wc = (unsigned short*)(W + off); off += 16384;   // 128x256 bf16
  float* colsum = W + off; off += HF;              // colsum + colsq contiguous
  float* colsq  = W + off; off += HF;
  float* scaleA = W + off; off += HF;
  float* shiftA = W + off; off += HF;
  float* gsum = W + off; off += (size_t)GG * 1536; // gsum + gcnt contiguous
  float* gcnt = W + off; off += 3 * GG;

  dim3 b256(256);
  k_zero<<<512, b256, 0, stream>>>(gsum, (long)GG * 1536 + 3 * GG);
  for (int t = 0; t < 3; ++t) {
    k_embed<<<(NN * EMBD + 255) / 256, b256, 0, stream>>>(feat[t], emb[t], hf[t], hbf[t], NN);
    k_count<<<(NN + 255) / 256, b256, 0, stream>>>(gid[t], gcnt + t * GG, NN);
  }

  for (int l = 0; l < 4; ++l) {
    int din = (l == 0) ? EMBD : HF;
    for (int t = 0; t < 3; ++t) {
      int pb = 15 + l * 18 + slot[t] * 6;
      const float* Wn = (const float*)d_in[pb + 0];
      const float* Ws = (const float*)d_in[pb + 1];
      const float* al = (const float*)d_in[pb + 2];
      const float* bb = (const float*)d_in[pb + 3];
      const float* be = (const float*)d_in[pb + 4];
      const float* ga = (const float*)d_in[pb + 5];

      k_zero<<<2048, b256, 0, stream>>>(msum, (long)NN * HF + NN);
      k_agg<<<(EE + 255) / 256, b256, 0, stream>>>(hf[t], src[t], dst[t], msum, deg, EE, din);
      k_mean_bf<<<(int)(((long)NN * din + 255) / 256), b256, 0, stream>>>(msum, deg, Mbf, NN, din);
      k_packw<<<(2 * din * HF + 255) / 256, b256, 0, stream>>>(Ws, Wn, Wc, din);
      k_zero<<<1, b256, 0, stream>>>(colsum, 256);
      if (din == EMBD)
        k_sage_gemm<EMBD><<<(NN + 63) / 64, b256, 0, stream>>>(
            hbf[t], Mbf, Wc, bb, al, colsum, colsq, hf[t], NN);
      else
        k_sage_gemm<HF><<<(NN + 63) / 64, b256, 0, stream>>>(
            hbf[t], Mbf, Wc, bb, al, colsum, colsq, hf[t], NN);
      k_bnfin<<<1, dim3(128), 0, stream>>>(colsum, colsq, ga, be, scaleA, shiftA, NN);
      k_bn_apply<<<(int)(((long)NN * HF + 255) / 256), b256, 0, stream>>>(
          hf[t], hbf[t], gid[t], gsum, scaleA, shiftA, NN, t * 512 + l * HF);
    }
  }
  k_final<<<(GG * 1536 + 255) / 256, b256, 0, stream>>>(gsum, gcnt, (float*)d_out);
}
